// HybridForecasterV6_83485574300333
// MI455X (gfx1250) — matrix operand order repaired and checked by reading, not compiled, not run on hardware
//
#include <hip/hip_runtime.h>
#include <hip/hip_bf16.h>

// ---------------------------------------------------------------------------
// HybridForecasterV6 on MI455X (gfx1250): WMMA f16 GEMMs for conv/GRU/attn,
// persistent-workgroup sequential scans with LDS atomics for the LIF/GRU
// recurrences, big-LDS (WGP 320KB) weight residency for the GRU.
// Round 2: 16x64 wave blocking (A-fragment reuse across 4 WMMAs) and
// transposed-V output so attention B fragments are contiguous b128 loads.
// ---------------------------------------------------------------------------

typedef __attribute__((ext_vector_type(16))) _Float16 v16h;
typedef __attribute__((ext_vector_type(8)))  float    v8f;
typedef _Float16 h16;

#define Bsz 32
#define Tt  512
#define INC 32
#define SNN 128
#define TS  4
#define HID 128
#define HOR 96
#define EPSV 1e-5f

// ---------------- WMMA fragment helpers (CDNA5 16x16x32 f16) ----------------
// A fragment: 16x32 f16, row-major source, row stride lda (elements).
// Lane L: m = L&15; holds K = (L>>4)*16 .. +15 (contiguous, matches load_frag_w).
__device__ __forceinline__ v16h load_frag_a(const h16* a, int lda) {
  const int lane = threadIdx.x & 31;
  const int m  = lane & 15;
  const int kb = (lane >> 4) * 16;
  const h16* row = a + (size_t)m * lda;
  v16h r;
#pragma unroll
  for (int j = 0; j < 16; ++j) r[j] = row[kb + j];
  return r;
}

// B fragment from W stored (N x Kd) row-major (i.e. computing A @ W^T).
// Lane L: n = L&15; holds K = (L>>4)*16 .. +15 -> 16 contiguous f16 of row n.
__device__ __forceinline__ v16h load_frag_w(const h16* w, int ldw) {
  const int lane = threadIdx.x & 31;
  const int n  = lane & 15;
  const int kb = (lane >> 4) * 16;
  const h16* row = w + (size_t)n * ldw + kb;
  v16h r;
#pragma unroll
  for (int j = 0; j < 16; ++j) r[j] = row[j];
  return r;
}

__device__ __forceinline__ v8f wmma16(v16h a, v16h b, v8f c) {
  return __builtin_amdgcn_wmma_f32_16x16x32_f16(false, a, false, b, (short)0, c,
                                                false, false);
}

// C 16x16 f32 store: lane L -> n = L&15, rows (L>>4)*8 + r.
__device__ __forceinline__ void store_frag_c(float* c, int ldc, v8f acc) {
  const int lane = threadIdx.x & 31;
  const int n  = lane & 15;
  const int mb = (lane >> 4) * 8;
#pragma unroll
  for (int r = 0; r < 8; ++r) c[(size_t)(mb + r) * ldc + n] = acc[r];
}

// ---------------- small utility kernels ----------------
__global__ void k_f32_to_f16(const float* __restrict__ s, h16* __restrict__ d, int n) {
  int i = blockIdx.x * blockDim.x + threadIdx.x;
  if (i < n) d[i] = (h16)s[i];
}
__global__ void k_zero_f32(float* p, int n) {
  int i = blockIdx.x * blockDim.x + threadIdx.x;
  if (i < n) p[i] = 0.f;
}
// im2col for conv1d K=3 pad=1: Xc[b*T+t][ci*3+k] = x[b, t+k-1, ci]
__global__ void k_im2col(const float* __restrict__ x, h16* __restrict__ Xc) {
  int i = blockIdx.x * blockDim.x + threadIdx.x;
  if (i >= Bsz * Tt * 96) return;
  int row = i / 96, col = i % 96;
  int b = row / Tt, t = row % Tt;
  int ci = col / 3, kk = col % 3;
  int tt = t + kk - 1;
  float v = (tt >= 0 && tt < Tt) ? x[((size_t)b * Tt + tt) * INC + ci] : 0.f;
  Xc[i] = (h16)v;
}
// hc columns 0..31 = x (f16)
__global__ void k_pack_hc_x(const float* __restrict__ x, h16* __restrict__ hc) {
  int i = blockIdx.x * blockDim.x + threadIdx.x;
  if (i >= Bsz * Tt * INC) return;
  int row = i >> 5, ci = i & 31;
  hc[(size_t)row * (INC + SNN) + ci] = (h16)x[i];
}

// ---- generic WMMA GEMM: C(MxN) = A(MxKd) @ W(NxKd)^T + bias ---------------
// Each wave computes a 16x64 strip: one A fragment feeds 4 WMMAs.
// Optional outputs: Cf (f32, MxN), Ch (f16, MxN), ChT (f16, NxM transposed).
__global__ __launch_bounds__(256)
void k_gemm_f16(const h16* __restrict__ A, const h16* __restrict__ W,
                const float* __restrict__ bias, float* __restrict__ Cf,
                h16* __restrict__ Ch, h16* __restrict__ ChT,
                int M, int N, int Kd) {
  int wave = (blockIdx.x * blockDim.x + threadIdx.x) >> 5;
  int tilesG = N >> 6;                 // groups of 4 N-tiles
  int waves  = (M >> 4) * tilesG;
  if (wave >= waves) return;
  int tm = wave / tilesG, tg = wave % tilesG;
  const h16* a0 = A + (size_t)tm * 16 * Kd;
  v8f acc[4] = {{}, {}, {}, {}};
  for (int k0 = 0; k0 < Kd; k0 += 32) {
    __builtin_prefetch(a0 + k0 + 64, 0, 1);  // global_prefetch_b8
    v16h af = load_frag_a(a0 + k0, Kd);
#pragma unroll
    for (int q = 0; q < 4; ++q) {
      const h16* w0 = W + (size_t)(tg * 64 + q * 16) * Kd + k0;
      acc[q] = wmma16(af, load_frag_w(w0, Kd), acc[q]);
    }
  }
  const int lane = threadIdx.x & 31;
  const int n = lane & 15, mb = (lane >> 4) * 8;
#pragma unroll
  for (int q = 0; q < 4; ++q) {
    int n0 = tg * 64 + q * 16;
    float bv = bias ? bias[n0 + n] : 0.f;
#pragma unroll
    for (int r = 0; r < 8; ++r) {
      float v = acc[q][r] + bv;
      int m = tm * 16 + mb + r;
      if (Cf)  Cf[(size_t)m * N + n0 + n] = v;
      if (Ch)  Ch[(size_t)m * N + n0 + n] = (h16)v;
      if (ChT) ChT[(size_t)(n0 + n) * M + m] = (h16)v;
    }
  }
}

// ---------------- BN1 stats: per-channel mean / inv-std over (B*T) ----------
__global__ __launch_bounds__(256)
void k_bn1_stats(const float* __restrict__ co, float* __restrict__ mean,
                 float* __restrict__ istd) {
  __shared__ float ss[256], sq[256];
  int c = blockIdx.x;  // 0..511
  float sm = 0.f, q = 0.f;
  for (int r = threadIdx.x; r < Bsz * Tt; r += 256) {
    float v = co[(size_t)r * (SNN * TS) + c];
    sm += v; q += v * v;
  }
  ss[threadIdx.x] = sm; sq[threadIdx.x] = q;
  __syncthreads();
  for (int st = 128; st > 0; st >>= 1) {
    if (threadIdx.x < (unsigned)st) { ss[threadIdx.x] += ss[threadIdx.x + st]; sq[threadIdx.x] += sq[threadIdx.x + st]; }
    __syncthreads();
  }
  if (threadIdx.x == 0) {
    float m = ss[0] / (float)(Bsz * Tt);
    float var = sq[0] / (float)(Bsz * Tt) - m * m;
    mean[c] = m; istd[c] = rsqrtf(var + EPSV);
  }
}

// ---------------- fused dual-LIF scan (persistent single workgroup) ---------
// 1024 threads = 32 waves; thread owns channel c = tid&127 for 4 batches.
__global__ __launch_bounds__(1024)
void k_lif(const float* __restrict__ co, const float* __restrict__ m1,
           const float* __restrict__ is1, const float* __restrict__ g1,
           const float* __restrict__ b1, const float* __restrict__ beta1,
           const float* __restrict__ g2, const float* __restrict__ b2,
           const float* __restrict__ beta2, h16* __restrict__ hc) {
  __shared__ float s1[SNN], s2v[SNN];
  const int tid = threadIdx.x;
  const int c = tid & 127, bq = tid >> 7;
  const float be1 = fminf(fmaxf(beta1[c], 0.f), 0.99f);
  const float be2 = fminf(fmaxf(beta2[c], 0.f), 0.99f);
  const float gg2 = g2[c], bb2 = b2[c];
  float mem1[4] = {0, 0, 0, 0}, rst1[4] = {0, 0, 0, 0};
  float mem2[4] = {0, 0, 0, 0}, rst2[4] = {0, 0, 0, 0};
  float acc[4] = {0, 0, 0, 0};
  for (int g = 0; g < Tt * TS; ++g) {
    int t = g >> 2, sub = g & 3, ch = sub * SNN + c;
    float mean = m1[ch], is = is1[ch], gg1 = g1[ch], bb1 = b1[ch];
    float spk[4], psum = 0.f, psq = 0.f;
#pragma unroll
    for (int j = 0; j < 4; ++j) {
      int b = bq * 4 + j;
      float v = co[((size_t)b * Tt + t) * (SNN * TS) + ch];
      float h = (v - mean) * is * gg1 + bb1;
      mem1[j] = be1 * mem1[j] + h - rst1[j];           // thr = 1.0
      float sp = (mem1[j] - 1.0f) > 0.f ? 1.f : 0.f;
      rst1[j] = sp; spk[j] = sp; psum += sp; psq += sp * sp;
    }
    if (tid < SNN) { s1[tid] = 0.f; s2v[tid] = 0.f; }
    __syncthreads();
    atomicAdd(&s1[c], psum);   // ds_add_f32
    atomicAdd(&s2v[c], psq);
    __syncthreads();
    float m2 = s1[c] * (1.f / Bsz);
    float var = s2v[c] * (1.f / Bsz) - m2 * m2;
    float is2 = rsqrtf(var + EPSV);
#pragma unroll
    for (int j = 0; j < 4; ++j) {
      float h2 = (spk[j] - m2) * is2 * gg2 + bb2;
      mem2[j] = be2 * mem2[j] + h2 - rst2[j];
      rst2[j] = (mem2[j] - 1.0f) > 0.f ? 1.f : 0.f;
      acc[j] += mem2[j];
    }
    if (sub == 3) {
#pragma unroll
      for (int j = 0; j < 4; ++j) {
        int b = bq * 4 + j;
        hc[((size_t)b * Tt + t) * (INC + SNN) + INC + c] = (h16)(acc[j] * 0.25f);
        acc[j] = 0.f;
      }
    }
    __syncthreads();
  }
}

// ---------------- persistent GRU: Whh resident in big LDS, WMMA per step ----
__global__ __launch_bounds__(256)
void k_gru(const h16* __restrict__ hc, const h16* __restrict__ Wih,
           const h16* __restrict__ WhhG, const float* __restrict__ bih,
           const float* __restrict__ bhh, h16* __restrict__ gout) {
  extern __shared__ unsigned char smem[];
  h16*   Whh = (h16*)smem;                                  // 384*128  (96KB)
  float* gi  = (float*)(smem + 98304);                      // 32*384   (48KB)
  float* gh  = (float*)(smem + 98304 + 49152);              // 32*384   (48KB)
  float* Hf  = (float*)(smem + 98304 + 98304);              // 32*128   (16KB)
  h16*   Ah  = (h16*)(smem + 98304 + 98304 + 16384);        // 32*128   (8KB)
  for (int i = threadIdx.x; i < 384 * HID; i += 256) Whh[i] = WhhG[i];
  for (int i = threadIdx.x; i < Bsz * HID; i += 256) { Hf[i] = 0.f; Ah[i] = (h16)0.f; }
  __syncthreads();
  const int wave = threadIdx.x >> 5;
  const int KIN = INC + SNN;  // 160
  for (int t = 0; t < Tt; ++t) {
    // gi = hc_t(32x160) @ Wih^T(160x384)
    for (int tile = wave; tile < 48; tile += 8) {
      int tm = tile / 24, tn = tile % 24;
      v8f acc = {};
      const h16* a0 = hc + ((size_t)(tm * 16) * Tt + t) * KIN;
      const h16* w0 = Wih + (size_t)tn * 16 * KIN;
#pragma unroll
      for (int k0 = 0; k0 < KIN; k0 += 32)
        acc = wmma16(load_frag_a(a0 + k0, Tt * KIN), load_frag_w(w0 + k0, KIN), acc);
      store_frag_c(gi + (size_t)(tm * 16) * 384 + tn * 16, 384, acc);
    }
    // gh = H(32x128) @ Whh^T(128x384), both operands in LDS
    for (int tile = wave; tile < 48; tile += 8) {
      int tm = tile / 24, tn = tile % 24;
      v8f acc = {};
      const h16* a0 = Ah + (size_t)(tm * 16) * HID;
      const h16* w0 = Whh + (size_t)tn * 16 * HID;
#pragma unroll
      for (int k0 = 0; k0 < HID; k0 += 32)
        acc = wmma16(load_frag_a(a0 + k0, HID), load_frag_w(w0 + k0, HID), acc);
      store_frag_c(gh + (size_t)(tm * 16) * 384 + tn * 16, 384, acc);
    }
    __syncthreads();
    // gates (PyTorch r,z,n order)
    for (int u = threadIdx.x; u < Bsz * HID; u += 256) {
      int b = u >> 7, j = u & 127;
      float ir = gi[b * 384 + j]       + bih[j];
      float iz = gi[b * 384 + 128 + j] + bih[128 + j];
      float in = gi[b * 384 + 256 + j] + bih[256 + j];
      float hr = gh[b * 384 + j]       + bhh[j];
      float hz = gh[b * 384 + 128 + j] + bhh[128 + j];
      float hn = gh[b * 384 + 256 + j] + bhh[256 + j];
      float r = 1.f / (1.f + __expf(-(ir + hr)));
      float z = 1.f / (1.f + __expf(-(iz + hz)));
      float n = tanhf(in + r * hn);
      float hnew = (1.f - z) * n + z * Hf[u];
      Hf[u] = hnew;
      gout[((size_t)b * Tt + t) * HID + j] = (h16)hnew;
    }
    __syncthreads();
    for (int u = threadIdx.x; u < Bsz * HID; u += 256) Ah[u] = (h16)Hf[u];
    __syncthreads();
  }
}

// ---------------- fused attention: sigmoid(QK^T*s)@V, column sums ----------
// One wave per 16-row Q tile. Q fragments hoisted (reused for 32 score
// tiles); V consumed through its transposed copy -> contiguous b128 loads.
__global__ __launch_bounds__(32)
void k_attn(const h16* __restrict__ Q, const h16* __restrict__ Km,
            const h16* __restrict__ Vt, float* __restrict__ out_sum) {
  __shared__ h16 S[16 * Tt];  // one 16-row score tile, f16 (16KB)
  const int b = blockIdx.x >> 5;
  const int rt = blockIdx.x & 31;
  const h16* q0 = Q + ((size_t)b * Tt + rt * 16) * HID;
  const float scale = 0.0883883476f;  // 128^-0.5
  const int lane = threadIdx.x & 31;
  const int n = lane & 15, mb = (lane >> 4) * 8;
  // hoist the 4 Q fragments (K = 128) for the whole row of score tiles
  v16h qf[4];
#pragma unroll
  for (int kq = 0; kq < 4; ++kq) qf[kq] = load_frag_a(q0 + kq * 32, HID);
  for (int nt = 0; nt < Tt / 16; ++nt) {
    v8f acc = {};
    const h16* k0p = Km + ((size_t)b * Tt + nt * 16) * HID;
#pragma unroll
    for (int kq = 0; kq < 4; ++kq)
      acc = wmma16(qf[kq], load_frag_w(k0p + kq * 32, HID), acc);
#pragma unroll
    for (int r = 0; r < 8; ++r) {
      float s = 1.f / (1.f + __expf(-acc[r] * scale));
      S[(size_t)(mb + r) * Tt + nt * 16 + n] = (h16)s;
    }
  }
  __syncthreads();
  // O(16x128) = S(16x512) @ V(512x128) via Vt (128 x B*T), 4-tile blocking
  for (int ng = 0; ng < 2; ++ng) {
    v8f acc[4] = {{}, {}, {}, {}};
    for (int k0 = 0; k0 < Tt; k0 += 32) {
      v16h af = load_frag_a(S + k0, Tt);
#pragma unroll
      for (int q = 0; q < 4; ++q) {
        const h16* w0 = Vt + (size_t)(ng * 64 + q * 16) * (Bsz * Tt) + (size_t)b * Tt + k0;
        acc[q] = wmma16(af, load_frag_w(w0, Bsz * Tt), acc[q]);
      }
    }
#pragma unroll
    for (int q = 0; q < 4; ++q) {
      float colsum = 0.f;
#pragma unroll
      for (int r = 0; r < 8; ++r) colsum += acc[q][r];
      atomicAdd(&out_sum[b * HID + ng * 64 + q * 16 + n], colsum);
    }
  }
}

// ---------------- BN3 over batch + head projection -------------------------
__global__ __launch_bounds__(128)
void k_head(const float* __restrict__ out_sum, const float* __restrict__ g3,
            const float* __restrict__ b3, const float* __restrict__ hw,
            const float* __restrict__ hb, float* __restrict__ out) {
  __shared__ float on[Bsz * HID];
  int c = threadIdx.x;  // 0..127
  float vals[Bsz];
  float m = 0.f;
  for (int b = 0; b < Bsz; ++b) { float v = out_sum[b * HID + c] * (1.f / Tt); vals[b] = v; m += v; }
  m *= (1.f / Bsz);
  float s2 = 0.f;
  for (int b = 0; b < Bsz; ++b) { float d = vals[b] - m; s2 += d * d; }
  float is = rsqrtf(s2 * (1.f / Bsz) + EPSV);
  for (int b = 0; b < Bsz; ++b) on[b * HID + c] = (vals[b] - m) * is * g3[c] + b3[c];
  __syncthreads();
  for (int o = threadIdx.x; o < Bsz * HOR; o += 128) {
    int b = o / HOR, hh = o % HOR;
    float acc = hb[hh];
    for (int cc = 0; cc < HID; ++cc) acc += on[b * HID + cc] * hw[hh * HID + cc];
    out[o] = acc;
  }
}

// ---------------------------------------------------------------------------
extern "C" void kernel_launch(void* const* d_in, const int* in_sizes, int n_in,
                              void* d_out, int out_size, void* d_ws, size_t ws_size,
                              hipStream_t stream) {
  const float* x      = (const float*)d_in[0];
  const float* conv_w = (const float*)d_in[1];
  const float* conv_b = (const float*)d_in[2];
  const float* bn1_g  = (const float*)d_in[3];
  const float* bn1_b  = (const float*)d_in[4];
  const float* beta1  = (const float*)d_in[5];
  const float* bn2_g  = (const float*)d_in[6];
  const float* bn2_b  = (const float*)d_in[7];
  const float* beta2  = (const float*)d_in[8];
  const float* gwih   = (const float*)d_in[9];
  const float* gwhh   = (const float*)d_in[10];
  const float* gbih   = (const float*)d_in[11];
  const float* gbhh   = (const float*)d_in[12];
  const float* q_w    = (const float*)d_in[13];
  const float* q_b    = (const float*)d_in[14];
  const float* k_w    = (const float*)d_in[15];
  const float* k_b    = (const float*)d_in[16];
  const float* v_w    = (const float*)d_in[17];
  const float* v_b    = (const float*)d_in[18];
  const float* bn3_g  = (const float*)d_in[19];
  const float* bn3_b  = (const float*)d_in[20];
  const float* head_w = (const float*)d_in[21];
  const float* head_b = (const float*)d_in[22];
  float* out = (float*)d_out;

  unsigned char* ws = (unsigned char*)d_ws;
  size_t off = 0;
  auto alloc = [&](size_t bytes) { void* p = ws + off; off += (bytes + 255) & ~size_t(255); return p; };
  h16*   Xc      = (h16*)alloc((size_t)Bsz * Tt * 96 * 2);
  h16*   Wc      = (h16*)alloc((size_t)512 * 96 * 2);
  float* conv_o  = (float*)alloc((size_t)Bsz * Tt * 512 * 4);
  float* bn1m    = (float*)alloc(512 * 4);
  float* bn1is   = (float*)alloc(512 * 4);
  h16*   hc      = (h16*)alloc((size_t)Bsz * Tt * (INC + SNN) * 2);
  h16*   Wih16   = (h16*)alloc((size_t)384 * 160 * 2);
  h16*   Whh16   = (h16*)alloc((size_t)384 * 128 * 2);
  h16*   g16     = (h16*)alloc((size_t)Bsz * Tt * HID * 2);
  h16*   qw16    = (h16*)alloc((size_t)HID * HID * 2);
  h16*   kw16    = (h16*)alloc((size_t)HID * HID * 2);
  h16*   vw16    = (h16*)alloc((size_t)HID * HID * 2);
  h16*   Qb      = (h16*)alloc((size_t)Bsz * Tt * HID * 2);
  h16*   Kb      = (h16*)alloc((size_t)Bsz * Tt * HID * 2);
  h16*   Vtb     = (h16*)alloc((size_t)HID * Bsz * Tt * 2);   // V transposed
  float* osum    = (float*)alloc((size_t)Bsz * HID * 4);

  auto cvt = [&](const float* s, h16* d, int n) {
    k_f32_to_f16<<<(n + 255) / 256, 256, 0, stream>>>(s, d, n);
  };
  cvt(conv_w, Wc, 512 * 96);
  cvt(gwih, Wih16, 384 * 160);
  cvt(gwhh, Whh16, 384 * 128);
  cvt(q_w, qw16, HID * HID);
  cvt(k_w, kw16, HID * HID);
  cvt(v_w, vw16, HID * HID);
  k_zero_f32<<<(Bsz * HID + 255) / 256, 256, 0, stream>>>(osum, Bsz * HID);

  {
    int n = Bsz * Tt * 96;
    k_im2col<<<(n + 255) / 256, 256, 0, stream>>>(x, Xc);
  }
  {  // conv as GEMM: (16384 x 512), K = 96; 16x64 strips -> 8192 waves
    int waves = (Bsz * Tt / 16) * (512 / 64);
    k_gemm_f16<<<(waves + 7) / 8, 256, 0, stream>>>(Xc, Wc, conv_b, conv_o,
                                                    nullptr, nullptr,
                                                    Bsz * Tt, 512, 96);
  }
  k_bn1_stats<<<512, 256, 0, stream>>>(conv_o, bn1m, bn1is);
  {
    int n = Bsz * Tt * INC;
    k_pack_hc_x<<<(n + 255) / 256, 256, 0, stream>>>(x, hc);
  }
  k_lif<<<1, 1024, 0, stream>>>(conv_o, bn1m, bn1is, bn1_g, bn1_b, beta1,
                                bn2_g, bn2_b, beta2, hc);
  {
    size_t smem = 98304 + 49152 + 49152 + 16384 + 8192;  // ~216KB of 320KB WGP LDS
    k_gru<<<1, 256, smem, stream>>>(hc, Wih16, Whh16, gbih, gbhh, g16);
  }
  {  // Q/K/V projections: (16384 x 128), K = 128; 16x64 strips -> 2048 waves
    int waves = (Bsz * Tt / 16) * (HID / 64);
    int blocks = (waves + 7) / 8;
    k_gemm_f16<<<blocks, 256, 0, stream>>>(g16, qw16, q_b, nullptr, Qb, nullptr,
                                           Bsz * Tt, HID, HID);
    k_gemm_f16<<<blocks, 256, 0, stream>>>(g16, kw16, k_b, nullptr, Kb, nullptr,
                                           Bsz * Tt, HID, HID);
    k_gemm_f16<<<blocks, 256, 0, stream>>>(g16, vw16, v_b, nullptr, nullptr, Vtb,
                                           Bsz * Tt, HID, HID);
  }
  k_attn<<<Bsz * (Tt / 16), 32, 0, stream>>>(Qb, Kb, Vtb, osum);
  k_head<<<1, 128, 0, stream>>>(osum, bn3_g, bn3_b, head_w, head_b, out);
}
